// DenseDeformableModule_63196148793820
// MI455X (gfx1250) — matrix operand order repaired and checked by reading, not compiled, not run on hardware
//
#include <hip/hip_runtime.h>
#include <hip/hip_bf16.h>

// ---------------- problem constants ----------------
#define B_    2
#define H_    192
#define W_    192
#define HW_   (H_ * W_)        // 36864
#define NPIX  (B_ * HW_)       // 73728
#define CMAX  256
#define OUTCH 192              // 64 (x) + 128 (comp)
#define PADH  8                // LDS row padding (halves) to break bank alignment

typedef __attribute__((ext_vector_type(16))) _Float16 v16h;
typedef __attribute__((ext_vector_type(8)))  _Float16 v8h;
typedef __attribute__((ext_vector_type(8)))  float    v8f;

// ---- WMMA f16 16x16x32 operand layouts (CDNA5 ISA 7.12.2) ----
// A 16x32 (MxK): lane L holds row M = L%16; with o = (L>=16 ? 16 : 0),
//   elems 0..15 -> K = o+0..o+15 (contiguous order chosen here, same as B; not ISA table order).
//   => packed K-major row: two contiguous 8-half (16B) vectors at +o and +o+8.
// B 32x16 (KxN): lane L holds col N = L%16; with q = (L>=16 ? 16 : 0),
//   elems 0..15 -> K = q+0..q+15 (contiguous order chosen here, same as A; not ISA table order).
//   => K-major row in LDS: two contiguous 8-half vectors at +q and +q+8.
// C/D 16x16 f32: lane L col N = L%16; elem v -> row M = v + (L>=16 ? 8 : 0).

union Frag16 { v16h v; v8h h[2]; };

__device__ __forceinline__ v16h load_a_frag(const _Float16* row_k) {
  // row_k already includes chunk base + per-lane o
  Frag16 u;
  u.h[0] = *(const v8h*)(row_k);
  u.h[1] = *(const v8h*)(row_k + 8);
  return u.v;
}
__device__ __forceinline__ v16h load_b_frag(const _Float16* row_k) {
  // row_k already includes chunk base + per-lane q
  Frag16 u;
  u.h[0] = *(const v8h*)(row_k);
  u.h[1] = *(const v8h*)(row_k + 8);
  return u.v;
}

// ---------------- utility kernels ----------------
__global__ void cast_f16_kernel(const float* __restrict__ s,
                                _Float16* __restrict__ d, int n) {
  int i = blockIdx.x * blockDim.x + threadIdx.x;
  if (i < n) d[i] = (_Float16)s[i];
}

// pack conv weights [ocSrc][cin][3][3] f32 -> [ocDst][tap*cin + c] f16,
// rows ocSrc..ocDst-1 zero-filled.
__global__ void pack_w_kernel(const float* __restrict__ w,
                              _Float16* __restrict__ wp,
                              int ocSrc, int ocDst, int cin) {
  int idx = blockIdx.x * blockDim.x + threadIdx.x;
  int n = ocDst * cin * 9;
  if (idx >= n) return;
  int o   = idx / (cin * 9);
  int rem = idx % (cin * 9);
  int c   = rem / 9;
  int tap = rem % 9;
  float v = (o < ocSrc) ? w[(o * cin + c) * 9 + tap] : 0.f;
  wp[o * (9 * cin) + tap * cin + c] = (_Float16)v;
}

__global__ void init_feat_kernel(const float* __restrict__ x,
                                 float* __restrict__ f32,
                                 _Float16* __restrict__ f16) {
  int i = blockIdx.x * blockDim.x + threadIdx.x;
  const int n = B_ * 64 * HW_;
  if (i >= n) return;
  int b = i / (64 * HW_);
  int c = (i / HW_) % 64;
  int r = i % HW_;
  float v = x[i];
  long di = (long)(b * CMAX + c) * HW_ + r;
  f32[di] = v;
  f16[di] = (_Float16)v;
}

__global__ void copy_x_kernel(const float* __restrict__ x,
                              float* __restrict__ out) {
  int i = blockIdx.x * blockDim.x + threadIdx.x;
  const int n = B_ * 64 * HW_;
  if (i >= n) return;
  int b = i / (64 * HW_);
  int c = (i / HW_) % 64;
  int r = i % HW_;
  out[(long)(b * OUTCH + c) * HW_ + r] = x[i];
}

// ---------------- 3x3 offset conv (LDS-staged implicit GEMM) ----------------
// Block: 128 threads = 4 waves; waves = 2 m-tiles x 2 n-tiles; 32 pixels/block.
__global__ __launch_bounds__(128) void off_conv_kernel(
    const _Float16* __restrict__ feat16, const _Float16* __restrict__ wpack,
    float* __restrict__ off, int cin) {
  __shared__ _Float16 smp[32 * (224 + PADH)];   // [p][k], k-major rows

  const int cinP = cin + PADH;
  int tid  = threadIdx.x;
  int lane = tid & 31;
  int wave = tid >> 5;
  int mtile = wave >> 1;
  int ntile = wave & 1;
  int ln = lane & 15;
  int ao = (lane >= 16) ? 16 : 0;
  int bo = (lane >= 16) ? 16 : 0;
  int pblk = blockIdx.x * 32;

  int n = pblk + ntile * 16 + ln;   // this lane's pixel (B column / D column)
  int b = n / HW_;
  int r = n % HW_;

  const _Float16* arow = wpack + (mtile * 16 + ln) * (9 * cin);
  const _Float16* brow = smp + (ntile * 16 + ln) * cinP;

  v8f acc = {};
  for (int tap = 0; tap < 9; ++tap) {
    int i = tap / 3, j = tap % 3;
    // cooperative fill: feat(c, y+i-1, x+j-1) for 32 pixels, zero OOB
    for (int idx = tid; idx < cin * 32; idx += 128) {
      int c = idx >> 5;
      int p = idx & 31;
      int np = pblk + p;
      int bb = np / HW_, rr = np % HW_;
      int yy = rr / W_ + i - 1;
      int xx = rr % W_ + j - 1;
      _Float16 v = (_Float16)0.f;
      if (yy >= 0 && yy < H_ && xx >= 0 && xx < W_)
        v = feat16[(long)(bb * CMAX + c) * HW_ + yy * W_ + xx];
      smp[p * cinP + c] = v;
    }
    __syncthreads();

    const _Float16* ap = arow + tap * cin + ao;
    for (int c0 = 0; c0 < cin; c0 += 32) {
      v16h a  = load_a_frag(ap + c0);
      v16h bm = load_b_frag(brow + c0 + bo);
      acc = __builtin_amdgcn_wmma_f32_16x16x32_f16(
          false, a, false, bm, (short)0, acc, false, false);
    }
    __syncthreads();
  }

  int hi = (lane >= 16) ? 8 : 0;
#pragma unroll
  for (int v = 0; v < 8; ++v) {
    int m2 = mtile * 16 + v + hi;
    if (m2 < 18) off[(long)(b * 18 + m2) * HW_ + r] = acc[v];
  }
}

// ---------------- deformable conv (bilinear gather -> LDS -> WMMA) -------
__global__ __launch_bounds__(128) void deform_kernel(
    float* feat32, _Float16* feat16,
    const float* __restrict__ off, const _Float16* __restrict__ wpack,
    int cin) {
  __shared__ _Float16 smp[32 * (224 + PADH)];   // [p][k]
  __shared__ float    sw[32][4];
  __shared__ int      sy0[32], sx0[32];

  const int cinP = cin + PADH;
  int tid  = threadIdx.x;
  int lane = tid & 31;
  int wave = tid >> 5;
  int mtile = wave >> 1;
  int ntile = wave & 1;
  int ln = lane & 15;
  int ao = (lane >= 16) ? 16 : 0;
  int bo = (lane >= 16) ? 16 : 0;
  int pblk = blockIdx.x * 32;

  int n = pblk + ntile * 16 + ln;
  int b = n / HW_;
  int r = n % HW_;

  const _Float16* arow = wpack + (mtile * 16 + ln) * (9 * cin);
  const _Float16* brow = smp + (ntile * 16 + ln) * cinP;

  v8f acc = {};
  for (int tap = 0; tap < 9; ++tap) {
    int i = tap / 3, j = tap % 3;
    if (tid < 32) {
      int np = pblk + tid;
      int bb = np / HW_, rr = np % HW_;
      int yy = rr / W_, xx = rr % W_;
      float dy = off[(long)(bb * 18 + 2 * tap)     * HW_ + rr];
      float dx = off[(long)(bb * 18 + 2 * tap + 1) * HW_ + rr];
      float ys = (float)(yy - 1 + i) + dy;
      float xs = (float)(xx - 1 + j) + dx;
      float y0f = floorf(ys), x0f = floorf(xs);
      float ty = ys - y0f, tx = xs - x0f;
      sy0[tid] = (int)y0f;
      sx0[tid] = (int)x0f;
      sw[tid][0] = (1.f - ty) * (1.f - tx);
      sw[tid][1] = (1.f - ty) * tx;
      sw[tid][2] = ty * (1.f - tx);
      sw[tid][3] = ty * tx;
    }
    __syncthreads();

    for (int idx = tid; idx < cin * 32; idx += 128) {
      int c = idx >> 5;
      int p = idx & 31;
      int np = pblk + p;
      int bb = np / HW_;
      const float* base = feat32 + (long)(bb * CMAX + c) * HW_;
      int y0 = sy0[p], x0 = sx0[p];
      float v = 0.f;
      int yc = y0,     xc = x0;
      if (yc >= 0 && yc < H_ && xc >= 0 && xc < W_) v += sw[p][0] * base[yc * W_ + xc];
      xc = x0 + 1;
      if (yc >= 0 && yc < H_ && xc >= 0 && xc < W_) v += sw[p][1] * base[yc * W_ + xc];
      yc = y0 + 1; xc = x0;
      if (yc >= 0 && yc < H_ && xc >= 0 && xc < W_) v += sw[p][2] * base[yc * W_ + xc];
      xc = x0 + 1;
      if (yc >= 0 && yc < H_ && xc >= 0 && xc < W_) v += sw[p][3] * base[yc * W_ + xc];
      smp[p * cinP + c] = (_Float16)v;
    }
    __syncthreads();

    const _Float16* ap = arow + tap * cin + ao;
    for (int c0 = 0; c0 < cin; c0 += 32) {
      v16h a  = load_a_frag(ap + c0);
      v16h bm = load_b_frag(brow + c0 + bo);
      acc = __builtin_amdgcn_wmma_f32_16x16x32_f16(
          false, a, false, bm, (short)0, acc, false, false);
    }
    __syncthreads();
  }

  int hi = (lane >= 16) ? 8 : 0;
#pragma unroll
  for (int v = 0; v < 8; ++v) {
    int m2 = mtile * 16 + v + hi;          // out ch 0..31
    long di = (long)(b * CMAX + cin + m2) * HW_ + r;
    feat32[di] = acc[v];
    feat16[di] = (_Float16)acc[v];
  }
}

// ---------------- 1x1 compression conv (LDS-staged WMMA GEMM) -------------
#define COMPP (256 + PADH)
__global__ __launch_bounds__(128) void comp_kernel(
    const _Float16* __restrict__ feat16, const _Float16* __restrict__ wc16,
    const float* __restrict__ bias, float* __restrict__ out) {
  __shared__ _Float16 smp[16 * COMPP];   // [p][k], k = 0..255

  int tid  = threadIdx.x;
  int lane = tid & 31;
  int wave = tid >> 5;
  int ln = lane & 15;
  int ao = (lane >= 16) ? 16 : 0;
  int bo = (lane >= 16) ? 16 : 0;
  int mtile = blockIdx.y * 4 + wave;       // 0..7 -> out ch 0..127
  int pblk = blockIdx.x * 16;

  int n = pblk + ln;
  int b = n / HW_;
  int r = n % HW_;

  // fill LDS: 256 channels x 16 pixels; consecutive lanes -> consecutive pixels
  for (int idx = tid; idx < 256 * 16; idx += 128) {
    int c = idx >> 4;
    int p = idx & 15;
    int np = pblk + p;
    int bb = np / HW_, rr = np % HW_;
    smp[p * COMPP + c] = feat16[(long)(bb * CMAX + c) * HW_ + rr];
  }
  __syncthreads();

  const _Float16* arow = wc16 + (mtile * 16 + ln) * 256 + ao;
  const _Float16* brow = smp + ln * COMPP + bo;

  v8f acc = {};
#pragma unroll
  for (int c0 = 0; c0 < 256; c0 += 32) {
    v16h a  = load_a_frag(arow + c0);
    v16h bm = load_b_frag(brow + c0);
    acc = __builtin_amdgcn_wmma_f32_16x16x32_f16(
        false, a, false, bm, (short)0, acc, false, false);
  }

  int hi = (lane >= 16) ? 8 : 0;
#pragma unroll
  for (int v = 0; v < 8; ++v) {
    int o = mtile * 16 + v + hi;           // 0..127
    out[(long)(b * OUTCH + 64 + o) * HW_ + r] = acc[v] + bias[o];
  }
}

// ---------------- host-side orchestration ----------------
extern "C" void kernel_launch(void* const* d_in, const int* in_sizes, int n_in,
                              void* d_out, int out_size, void* d_ws, size_t ws_size,
                              hipStream_t stream) {
  // setup_inputs() dict order: x, (w_off0, w_def0), ..., (w_off5, w_def5),
  // w_comp, b_comp
  const float* x = (const float*)d_in[0];
  const float* w_off[6];
  const float* w_def[6];
  for (int l = 0; l < 6; ++l) {
    w_off[l] = (const float*)d_in[1 + 2 * l];
    w_def[l] = (const float*)d_in[2 + 2 * l];
  }
  const float* w_comp = (const float*)d_in[13];
  const float* b_comp = (const float*)d_in[14];
  float* out = (float*)d_out;

  // carve workspace
  char* p = (char*)d_ws;
  auto carve = [&](size_t bytes) -> char* {
    char* q = p;
    p += (bytes + 255) & ~(size_t)255;
    return q;
  };
  float*    feat32 = (float*)   carve((size_t)B_ * CMAX * HW_ * sizeof(float));
  _Float16* feat16 = (_Float16*)carve((size_t)B_ * CMAX * HW_ * sizeof(_Float16));
  float*    offbuf = (float*)   carve((size_t)B_ * 18 * HW_ * sizeof(float));
  _Float16* wo16[6];
  _Float16* wd16[6];
  for (int l = 0; l < 6; ++l) {
    int cin = 64 + 32 * l;
    wo16[l] = (_Float16*)carve((size_t)32 * 9 * cin * sizeof(_Float16)); // padded to 32 rows
    wd16[l] = (_Float16*)carve((size_t)32 * 9 * cin * sizeof(_Float16));
  }
  _Float16* wc16 = (_Float16*)carve((size_t)128 * 256 * sizeof(_Float16));

  // pack/cast weights
  for (int l = 0; l < 6; ++l) {
    int cin = 64 + 32 * l;
    int n1 = 32 * cin * 9;
    pack_w_kernel<<<(n1 + 255) / 256, 256, 0, stream>>>(w_off[l], wo16[l], 18, 32, cin);
    pack_w_kernel<<<(n1 + 255) / 256, 256, 0, stream>>>(w_def[l], wd16[l], 32, 32, cin);
  }
  cast_f16_kernel<<<(128 * 256 + 255) / 256, 256, 0, stream>>>(w_comp, wc16, 128 * 256);

  // init feat (f32 + f16) and x-passthrough output channels
  const int nfi = B_ * 64 * HW_;
  init_feat_kernel<<<(nfi + 255) / 256, 256, 0, stream>>>(x, feat32, feat16);
  copy_x_kernel<<<(nfi + 255) / 256, 256, 0, stream>>>(x, out);

  // 6 dense deformable layers
  for (int l = 0; l < 6; ++l) {
    int cin = 64 + 32 * l;
    off_conv_kernel<<<NPIX / 32, 128, 0, stream>>>(feat16, wo16[l], offbuf, cin);
    deform_kernel<<<NPIX / 32, 128, 0, stream>>>(feat32, feat16, offbuf, wd16[l], cin);
  }

  // 1x1 compression + bias -> output channels 64..191
  comp_kernel<<<dim3(NPIX / 16, 2), 128, 0, stream>>>(feat16, wc16, b_comp, out);
}
